// SimpleBEV_2628519985749
// MI455X (gfx1250) — matrix operand order repaired and checked by reading, not compiled, not run on hardware
//
#include <hip/hip_runtime.h>
#include <hip/hip_bf16.h>

// ---------------------------------------------------------------------------
// SimpleBEV on gfx1250: project -> bilinear -> multi-cam mean -> (implicit
// im2col) WMMA f16 GEMM for the 3x3 conv (K=9216) with async global->LDS
// double-buffered staging -> instance-norm + GELU.
// ---------------------------------------------------------------------------

typedef _Float16 half8  __attribute__((ext_vector_type(8)));
typedef _Float16 half16 __attribute__((ext_vector_type(16)));
typedef float    float8 __attribute__((ext_vector_type(8)));

#define ZDIM 128
#define YDIM 8
#define XDIM 128
#define CDIM 128
#define HF 32
#define WF 88
#define NCAM 6
#define CIN 1024              // C*Y
#define NPIX (ZDIM*XDIM)      // 16384
#define KTOT (CIN*9)          // 9216
#define KSTEPS (KTOT/32)      // 288
#define PADW 130              // zero-padded spatial extent (128 + 1 ring)
#define FPAD_HALVES ((size_t)PADW*PADW*CIN)   // 17,305,600 halves

__device__ __forceinline__ int iclamp(int v, int lo, int hi) {
    return v < lo ? lo : (v > hi ? hi : v);
}

// ---------------------------------------------------------------------------
// Kernel 1: per-camera pixB_T_cam0 (3x4).  camX_T_cam0 = (Rs^T*R0 | Rs^T(t0-ts))
// pix = K_scaled * camX_T_cam0 ; row2 of pix == row2 of camX_T_cam0 (z_cam).
// ---------------------------------------------------------------------------
__global__ void bev_setup_mats(const float* __restrict__ c2e,
                               const float* __restrict__ intr,
                               float* __restrict__ mats) {
    int s = threadIdx.x;
    if (s >= NCAM) return;
    float R0[3][3], t0[3], Rs[3][3], ts[3];
    #pragma unroll
    for (int r = 0; r < 3; ++r) {
        #pragma unroll
        for (int c = 0; c < 3; ++c) {
            R0[r][c] = c2e[r * 4 + c];
            Rs[r][c] = c2e[s * 16 + r * 4 + c];
        }
        t0[r] = c2e[r * 4 + 3];
        ts[r] = c2e[s * 16 + r * 4 + 3];
    }
    float Rb[3][3], tb[3];
    #pragma unroll
    for (int r = 0; r < 3; ++r) {
        #pragma unroll
        for (int c = 0; c < 3; ++c)
            Rb[r][c] = Rs[0][r] * R0[0][c] + Rs[1][r] * R0[1][c] + Rs[2][r] * R0[2][c];
        tb[r] = Rs[0][r] * (t0[0] - ts[0]) + Rs[1][r] * (t0[1] - ts[1]) + Rs[2][r] * (t0[2] - ts[2]);
    }
    const float sxs = (float)WF / 704.0f * 0.44f;
    const float sys = (float)HF / 256.0f * 0.284f;
    float fx = intr[s * 16 + 0] * sxs;
    float fy = intr[s * 16 + 5] * sys;
    float cx = intr[s * 16 + 2] * sxs;
    float cy = intr[s * 16 + 6] * sys;
    float* P = mats + s * 12;
    #pragma unroll
    for (int c = 0; c < 3; ++c) {
        P[0 + c] = fx * Rb[0][c] + cx * Rb[2][c];
        P[4 + c] = fy * Rb[1][c] + cy * Rb[2][c];
        P[8 + c] = Rb[2][c];
    }
    P[3]  = fx * tb[0] + cx * tb[2];
    P[7]  = fy * tb[1] + cy * tb[2];
    P[11] = tb[2];
}

// ---------------------------------------------------------------------------
// Kernel 2: zero-fill the padded BEV feature map (16B vector stores).
// ---------------------------------------------------------------------------
__global__ void bev_zero(half8* __restrict__ p, int n8) {
    half8 zv;
    #pragma unroll
    for (int q = 0; q < 8; ++q) zv[q] = (_Float16)0.0f;
    for (int i = blockIdx.x * blockDim.x + threadIdx.x; i < n8; i += gridDim.x * blockDim.x)
        p[i] = zv;
}

// ---------------------------------------------------------------------------
// Kernel 3: reorder conv weights (oc,ic,ky,kx) f32 -> w16[oc][tap*1024+ic] f16
// ---------------------------------------------------------------------------
__global__ void bev_wconv(const float* __restrict__ w, _Float16* __restrict__ w16) {
    int i = blockIdx.x * 256 + threadIdx.x;
    if (i >= CDIM * CIN * 9) return;
    int oc  = i / (CIN * 9);
    int rem = i - oc * (CIN * 9);
    int ic  = rem / 9;
    int tap = rem - ic * 9;
    w16[oc * KTOT + tap * CIN + ic] = (_Float16)w[i];
}

// ---------------------------------------------------------------------------
// Kernel 4: projection + bilinear + multi-camera masked mean, written into the
// zero-padded pixel-major map fpad[(z+1)][(x+1)][cin] f16 (cin = c*8 + y).
// ---------------------------------------------------------------------------
__global__ __launch_bounds__(256) void bev_project(const float* __restrict__ feat,
                                                   const float* __restrict__ mats,
                                                   _Float16* __restrict__ fpad) {
    int n = blockIdx.x * 256 + threadIdx.x;       // 131072 voxels
    int z = n >> 10;
    int rem = n & 1023;
    int y = rem >> 7;
    int x = rem & 127;

    float wx = (float)x * 0.78125f - 49.609375f;  // x*vx + xmin + vx/2
    float wy = (float)y * 1.25f    - 3.375f;      // y*vy + ymin + vy/2
    float wz = (float)z * 0.78125f - 49.609375f;

    int   offs[NCAM][4];
    float wts[NCAM][4];
    bool  vld[NCAM];

    #pragma unroll
    for (int s = 0; s < NCAM; ++s) {
        const float* P = mats + s * 12;
        float px_ = P[0] * wx + P[1] * wy + P[2]  * wz + P[3];
        float py_ = P[4] * wx + P[5] * wy + P[6]  * wz + P[7];
        float pz_ = P[8] * wx + P[9] * wy + P[10] * wz + P[11];
        float depth = fmaxf(pz_, 1e-4f);
        float xp = px_ / depth;
        float yp = py_ / depth;
        vld[s] = (xp > -0.5f) && (xp < (float)WF - 0.5f) &&
                 (yp > -0.5f) && (yp < (float)HF - 0.5f) && (pz_ > 0.0f);
        float gx = fminf(fmaxf(xp * (2.0f / (WF - 1)) - 1.0f, -2.0f), 2.0f);
        float gy = fminf(fmaxf(yp * (2.0f / (HF - 1)) - 1.0f, -2.0f), 2.0f);
        float pxx = ((gx + 1.0f) * (float)WF - 1.0f) * 0.5f;
        float pyy = ((gy + 1.0f) * (float)HF - 1.0f) * 0.5f;
        float fx0 = floorf(pxx), fy0 = floorf(pyy);
        float dx = pxx - fx0, dy = pyy - fy0;
        int x0 = (int)fx0, y0 = (int)fy0;
        int x1 = x0 + 1,   y1 = y0 + 1;
        float bx0 = (x0 >= 0 && x0 < WF) ? 1.0f : 0.0f;
        float bx1 = (x1 >= 0 && x1 < WF) ? 1.0f : 0.0f;
        float by0 = (y0 >= 0 && y0 < HF) ? 1.0f : 0.0f;
        float by1 = (y1 >= 0 && y1 < HF) ? 1.0f : 0.0f;
        int cx0 = iclamp(x0, 0, WF - 1), cx1 = iclamp(x1, 0, WF - 1);
        int cy0 = iclamp(y0, 0, HF - 1), cy1 = iclamp(y1, 0, HF - 1);
        offs[s][0] = cy0 * WF + cx0;  wts[s][0] = (1.0f - dx) * (1.0f - dy) * bx0 * by0;
        offs[s][1] = cy0 * WF + cx1;  wts[s][1] = dx * (1.0f - dy) * bx1 * by0;
        offs[s][2] = cy1 * WF + cx0;  wts[s][2] = (1.0f - dx) * dy * bx0 * by1;
        offs[s][3] = cy1 * WF + cx1;  wts[s][3] = dx * dy * bx1 * by1;
    }

    _Float16* outp = fpad + ((size_t)((z + 1) * PADW + (x + 1))) * CIN + y;
    for (int c = 0; c < CDIM; ++c) {
        float num = 0.0f, den = 0.0f;
        #pragma unroll
        for (int s = 0; s < NCAM; ++s) {
            if (vld[s]) {
                const float* base = feat + (s * CDIM + c) * (HF * WF);
                float v = wts[s][0] * base[offs[s][0]] + wts[s][1] * base[offs[s][1]] +
                          wts[s][2] * base[offs[s][2]] + wts[s][3] * base[offs[s][3]];
                if (v != 0.0f) { num += v; den += 1.0f; }
            }
        }
        outp[c << 3] = (_Float16)(num / (1e-6f + den));
    }
}

// ---------------------------------------------------------------------------
// Kernel 5: implicit-im2col GEMM conv via V_WMMA_F32_16X16X32_F16.
// Block = 256 threads (8 waves). Block covers 64 pixels; wave w owns out
// channels [16w,16w+16) with 4 accumulator tiles (M16 x N64).  B tiles are
// double-buffered in LDS and staged with GLOBAL_LOAD_ASYNC_TO_LDS_B128
// (ASYNCcnt), overlapping the DMA of tile k+1 with the WMMAs of tile k.
// ---------------------------------------------------------------------------
__global__ __launch_bounds__(256) void bev_conv_gemm(const _Float16* __restrict__ w16,
                                                     const _Float16* __restrict__ fpad,
                                                     float* __restrict__ cout) {
    __shared__ __align__(16) _Float16 Bt[2][64 * 32];   // double buffer, 2 x 4 KB

    const int tid   = threadIdx.x;
    const int wave  = tid >> 5;
    const int lane  = tid & 31;
    const int lhalf = lane >> 4;       // 0: lanes 0-15, 1: lanes 16-31
    const int l15   = lane & 15;
    const int pixBase = blockIdx.x * 64;
    const int ocBase  = wave * 16;

    float8 acc[4];
    #pragma unroll
    for (int nt = 0; nt < 4; ++nt)
        #pragma unroll
        for (int j = 0; j < 8; ++j) acc[nt][j] = 0.0f;

    // A-fragment base: row M = ocBase + l15; lane group K offset = lhalf*8
    const _Float16* aPtr = w16 + (ocBase + l15) * KTOT + lhalf * 8;

    // B staging: one async b128 per thread per K-chunk (64 px * 32 k halves)
    const int bn = tid >> 2;            // 0..63 pixel within tile
    const int bk = (tid & 3) * 8;       // 0..24 k within 32-chunk
    const int bz = (pixBase + bn) >> 7;
    const int bx = (pixBase + bn) & 127;
    const unsigned ldsDst0 = (unsigned)(unsigned long long)&Bt[0][bn * 32 + bk];
    const unsigned ldsDst1 = (unsigned)(unsigned long long)&Bt[1][bn * 32 + bk];

    // stage K-chunk `ks` into buffer ks&1 (padded map -> no bounds check)
    #define STAGE(ks)                                                           \
        do {                                                                    \
            int _kb  = (ks) * 32;                                               \
            int _tap = _kb >> 10;                                               \
            int _icb = _kb & 1023;                                              \
            const _Float16* _g = fpad +                                         \
                (size_t)((bz + _tap / 3) * PADW + bx + _tap % 3) * CIN + _icb + bk; \
            unsigned _d = ((ks) & 1) ? ldsDst1 : ldsDst0;                       \
            asm volatile("global_load_async_to_lds_b128 %0, %1, off"            \
                         :: "v"(_d), "v"(_g) : "memory");                       \
        } while (0)

    STAGE(0);
    asm volatile("s_wait_asynccnt 0" ::: "memory");
    __syncthreads();

    for (int ks = 0; ks < KSTEPS; ++ks) {
        if (ks + 1 < KSTEPS) STAGE(ks + 1);       // DMA next tile during compute

        // A fragment per documented 16-bit 16x32 layout:
        //   lanes 0-15:  h0-7 -> K 0-7,  h8-15 -> K 16-23
        //   lanes 16-31: h0-7 -> K 8-15, h8-15 -> K 24-31
        const int kbase = ks * 32;
        half8 alo = *(const half8*)(aPtr + kbase);
        half8 ahi = *(const half8*)(aPtr + kbase + 16);
        __builtin_prefetch(aPtr + kbase + 32, 0, 0);
        half16 a;
        #pragma unroll
        for (int q = 0; q < 8; ++q) { a[q] = alo[q]; a[q + 8] = ahi[q]; }

        const _Float16* Bbuf = Bt[ks & 1];
        #pragma unroll
        for (int nt = 0; nt < 4; ++nt) {
            // B fragment: lane N = l15 (+16*nt), h0-7 -> K = 8*lhalf + h, h8-15 -> K = 16 + 8*lhalf + (h-8)
            const _Float16* bp = Bbuf + (nt * 16 + l15) * 32 + lhalf * 8;
            half8 blo = *(const half8*)bp;
            half8 bhi = *(const half8*)(bp + 16);
            half16 b;
            #pragma unroll
            for (int q = 0; q < 8; ++q) { b[q] = blo[q]; b[q + 8] = bhi[q]; }
            acc[nt] = __builtin_amdgcn_wmma_f32_16x16x32_f16(
                false, a, false, b, (short)0, acc[nt], false, false);
        }

        // next tile resident + everyone done reading current buffer
        asm volatile("s_wait_asynccnt 0" ::: "memory");
        __syncthreads();
    }
    #undef STAGE

    // C/D layout: VGPR j, lanes 0-15: M=j, N=lane; lanes 16-31: M=j+8, N=lane-16
    #pragma unroll
    for (int nt = 0; nt < 4; ++nt) {
        const int pix = pixBase + nt * 16 + l15;
        #pragma unroll
        for (int j = 0; j < 8; ++j) {
            const int oc = ocBase + j + 8 * lhalf;
            cout[oc * NPIX + pix] = acc[nt][j];
        }
    }
}

// ---------------------------------------------------------------------------
// Kernel 6: per-out-channel instance norm + exact GELU.
// ---------------------------------------------------------------------------
__global__ __launch_bounds__(256) void bev_norm_gelu(const float* __restrict__ cout,
                                                     float* __restrict__ out) {
    const int oc = blockIdx.x;
    const float* p = cout + oc * NPIX;
    float s = 0.0f, s2 = 0.0f;
    for (int i = threadIdx.x; i < NPIX; i += 256) {
        float v = p[i];
        s += v;
        s2 += v * v;
    }
    __shared__ float sh0[256], sh1[256];
    sh0[threadIdx.x] = s;
    sh1[threadIdx.x] = s2;
    __syncthreads();
    for (int off = 128; off > 0; off >>= 1) {
        if (threadIdx.x < off) {
            sh0[threadIdx.x] += sh0[threadIdx.x + off];
            sh1[threadIdx.x] += sh1[threadIdx.x + off];
        }
        __syncthreads();
    }
    const float mu   = sh0[0] * (1.0f / NPIX);
    const float var  = sh1[0] * (1.0f / NPIX) - mu * mu;
    const float rstd = rsqrtf(var + 1e-5f);
    for (int i = threadIdx.x; i < NPIX; i += 256) {
        float v = (p[i] - mu) * rstd;
        out[oc * NPIX + i] = 0.5f * v * (1.0f + erff(v * 0.70710678118654752f));
    }
}

// ---------------------------------------------------------------------------
extern "C" void kernel_launch(void* const* d_in, const int* in_sizes, int n_in,
                              void* d_out, int out_size, void* d_ws, size_t ws_size,
                              hipStream_t stream) {
    const float* cam_feat = (const float*)d_in[0];   // (1,6,128,32,88)
    const float* cam2ego  = (const float*)d_in[1];   // (1,6,4,4)
    const float* intr     = (const float*)d_in[2];   // (1,6,4,4)
    const float* conv_w   = (const float*)d_in[3];   // (128,1024,3,3)
    float* out = (float*)d_out;                      // (1,128,128,128)

    char* ws = (char*)d_ws;
    float*     mats = (float*)ws;                                  //   1 KB
    _Float16*  fpad = (_Float16*)(ws + 1024);                      //  34.6 MB padded
    _Float16*  w16  = (_Float16*)(ws + 1024 + FPAD_HALVES * 2);    //   2.4 MB
    float*     cwrk = (float*)(ws + 1024 + FPAD_HALVES * 2
                               + (size_t)CDIM * KTOT * 2);         //   8.4 MB

    bev_setup_mats<<<1, 32, 0, stream>>>(cam2ego, intr, mats);
    bev_zero<<<2048, 256, 0, stream>>>((half8*)fpad, (int)(FPAD_HALVES / 8));
    bev_wconv<<<(CDIM * CIN * 9 + 255) / 256, 256, 0, stream>>>(conv_w, w16);
    bev_project<<<(ZDIM * YDIM * XDIM) / 256, 256, 0, stream>>>(cam_feat, mats, fpad);
    bev_conv_gemm<<<NPIX / 64, 256, 0, stream>>>(w16, fpad, cwrk);
    bev_norm_gelu<<<CDIM, 256, 0, stream>>>(cwrk, out);
}
